// BlankAttention_27341761806384
// MI455X (gfx1250) — matrix operand order repaired and checked by reading, not compiled, not run on hardware
//
#include <hip/hip_runtime.h>

// Problem constants (match reference)
#define BB   2
#define SS   2048
#define DM   1024
#define HH   16
#define DH   64
#define NQKV 3072            // 3*H*D
#define MTOT (BB * SS)       // 4096

typedef __attribute__((ext_vector_type(16))) __bf16       v16bf;
typedef __attribute__((ext_vector_type(8)))  float        v8f;
typedef __attribute__((ext_vector_type(4)))  unsigned int uv4;

union FragU {
    v16bf          v;
    uv4            q[2];
    unsigned short s[16];
};

// fp32 -> bf16, round-to-nearest-even
__device__ __forceinline__ unsigned short f2bf(float f) {
    unsigned int u = __float_as_uint(f);
    u += 0x7FFFu + ((u >> 16) & 1u);
    return (unsigned short)(u >> 16);
}

// A-fragment 16x32 bf16 (row-major source, ld elements per row).
// Lanes 0-15: row M=lane, K chunks {k..k+7, k+16..k+23}
// Lanes 16-31: row M=lane-16, K chunks {k+8..k+15, k+24..k+31}
__device__ __forceinline__ v16bf load_frag_a(const unsigned short* __restrict__ base,
                                             int row_base, int ld, int k_base) {
    int lane = threadIdx.x & 31;
    int hi   = lane >> 4;
    const unsigned short* p =
        base + (size_t)(row_base + (lane & 15)) * ld + k_base + hi * 8;
    FragU f;
    f.q[0] = *(const uv4*)(p);
    f.q[1] = *(const uv4*)(p + 16);
    return f.v;
}

// B-fragment 32x16 bf16. Source laid out column-contiguous: element (n,k) at
// base + n*ld + k. Lanes 0-15: col N=lane, K {k..k+7, k+16..k+23}; lanes 16-31: K {k+8..k+15, k+24..k+31}.
__device__ __forceinline__ v16bf load_frag_b(const unsigned short* __restrict__ base,
                                             int n_base, int k_base, int ld) {
    int lane = threadIdx.x & 31;
    int hi   = lane >> 4;
    const unsigned short* p =
        base + (size_t)(n_base + (lane & 15)) * ld + k_base + hi * 8;
    FragU f;
    f.q[0] = *(const uv4*)(p);
    f.q[1] = *(const uv4*)(p + 16);
    return f.v;
}

__device__ __forceinline__ v8f wmma_bf16(v16bf a, v16bf b, v8f c) {
    return __builtin_amdgcn_wmma_f32_16x16x32_bf16(
        false, a, false, b, (short)0, c, false, false);
}

// ---------------- pack kernels ----------------

__global__ void k_pack_x(const float* __restrict__ x,
                         unsigned short* __restrict__ xb, int n) {
    int i = blockIdx.x * blockDim.x + threadIdx.x;
    if (i < n) xb[i] = f2bf(x[i]);
}

// wt[n*rows + k] = bf16(w[k*cols + n])   (rows = K, cols = N)
__global__ void k_pack_wT(const float* __restrict__ w,
                          unsigned short* __restrict__ wt,
                          int rows, int cols) {
    int i = blockIdx.x * blockDim.x + threadIdx.x;
    if (i >= rows * cols) return;
    int n = i / rows;
    int k = i - n * rows;
    wt[i] = f2bf(w[(size_t)k * cols + n]);
}

// ---------------- QKV projection GEMM ----------------
// C[m][n] = sum_k xb[m][k] * WqkvT[n][k];  M=4096, N=3072, K=1024
// One wave per 32x64 tile (2 A-frags x 4 B-frags -> 8 WMMA per K-step);
// scatter results into q[S][D], k[S][D], vT[D][S] per head.
__global__ __launch_bounds__(32)
void k_gemm_qkv(const unsigned short* __restrict__ xb,
                const unsigned short* __restrict__ wqkvT,
                unsigned short* __restrict__ qb,
                unsigned short* __restrict__ kb,
                unsigned short* __restrict__ vb) {
    int nt = blockIdx.x;          // N/64 tiles
    int mt = blockIdx.y;          // M/32 tiles
    int n0 = nt * 64;
    int m0 = mt * 32;
    v8f acc[2][4] = {};
    for (int k = 0; k < DM; k += 32) {
        v16bf a0 = load_frag_a(xb, m0,      DM, k);
        v16bf a1 = load_frag_a(xb, m0 + 16, DM, k);
        v16bf b0 = load_frag_b(wqkvT, n0,      k, DM);
        v16bf b1 = load_frag_b(wqkvT, n0 + 16, k, DM);
        v16bf b2 = load_frag_b(wqkvT, n0 + 32, k, DM);
        v16bf b3 = load_frag_b(wqkvT, n0 + 48, k, DM);
        acc[0][0] = wmma_bf16(a0, b0, acc[0][0]);
        acc[0][1] = wmma_bf16(a0, b1, acc[0][1]);
        acc[0][2] = wmma_bf16(a0, b2, acc[0][2]);
        acc[0][3] = wmma_bf16(a0, b3, acc[0][3]);
        acc[1][0] = wmma_bf16(a1, b0, acc[1][0]);
        acc[1][1] = wmma_bf16(a1, b1, acc[1][1]);
        acc[1][2] = wmma_bf16(a1, b2, acc[1][2]);
        acc[1][3] = wmma_bf16(a1, b3, acc[1][3]);
    }
    int lane   = threadIdx.x & 31;
    int cl     = lane & 15;
    int rowoff = (lane >> 4) * 8;
#pragma unroll
    for (int t = 0; t < 2; ++t) {
#pragma unroll
        for (int c = 0; c < 4; ++c) {
            int n  = n0 + c * 16 + cl;
            int h  = n / 192;
            int cc = n - h * 192;           // 0..191: q | k | v within head
#pragma unroll
            for (int r = 0; r < 8; ++r) {
                int m    = m0 + t * 16 + r + rowoff;
                int bidx = m >> 11;         // m / S
                int s    = m & (SS - 1);
                unsigned short val = f2bf(acc[t][c][r]);
                size_t head = (size_t)(bidx * HH + h);
                if (cc < 64)
                    qb[(head * SS + s) * DH + cc] = val;
                else if (cc < 128)
                    kb[(head * SS + s) * DH + (cc - 64)] = val;
                else
                    vb[(head * DH + (cc - 128)) * SS + s] = val;   // V transposed
            }
        }
    }
}

// ---------------- flash attention ----------------
// One wave per (b*H+h, 16-query tile). Key blocks of 32, online softmax.
__global__ __launch_bounds__(32)
void k_attn(const unsigned short* __restrict__ qb,
            const unsigned short* __restrict__ kb,
            const unsigned short* __restrict__ vb,
            unsigned short* __restrict__ ao) {
    int qt = blockIdx.x;                      // S/16 tiles
    int bh = blockIdx.y;                      // B*H heads
    const unsigned short* q  = qb + (size_t)bh * SS * DH;
    const unsigned short* kk = kb + (size_t)bh * SS * DH;
    const unsigned short* vv = vb + (size_t)bh * DH * SS;
    int q0 = qt * 16;

    v16bf qa0 = load_frag_a(q, q0, DH, 0);
    v16bf qa1 = load_frag_a(q, q0, DH, 32);

    int lane   = threadIdx.x & 31;
    int cl     = lane & 15;
    int rowoff = (lane >> 4) * 8;

    float mi[8], li[8];
    v8f o0 = {}, o1 = {}, o2 = {}, o3 = {};
#pragma unroll
    for (int r = 0; r < 8; ++r) { mi[r] = -3.0e38f; li[r] = 0.0f; }

    __shared__ float lds[16][34];

    int nb = q0 / 32 + 1;                     // causal: key blocks touching tril
    for (int blk = 0; blk < nb; ++blk) {
        int j = blk * 32;
        // S_tile = Q (16xD) @ K^T (Dx32): two 16x16 C-fragments
        v8f s0 = {}, s1 = {};
        v16bf kb00 = load_frag_b(kk, j,      0,  DH);
        v16bf kb01 = load_frag_b(kk, j,      32, DH);
        s0 = wmma_bf16(qa0, kb00, s0);
        s0 = wmma_bf16(qa1, kb01, s0);
        v16bf kb10 = load_frag_b(kk, j + 16, 0,  DH);
        v16bf kb11 = load_frag_b(kk, j + 16, 32, DH);
        s1 = wmma_bf16(qa0, kb10, s1);
        s1 = wmma_bf16(qa1, kb11, s1);

        bool last = (blk == nb - 1);          // only the diagonal block masks
#pragma unroll
        for (int r = 0; r < 8; ++r) {
            int   query = q0 + r + rowoff;
            float x0 = s0[r] * 0.125f;        // 1/sqrt(64)
            float x1 = s1[r] * 0.125f;
            if (last) {
                if (j + cl > query)      x0 = -3.0e38f;
                if (j + 16 + cl > query) x1 = -3.0e38f;
            }
            // row-max across the 16-lane group holding this row
            float mx = fmaxf(x0, x1);
            mx = fmaxf(mx, __shfl_xor(mx, 1, 32));
            mx = fmaxf(mx, __shfl_xor(mx, 2, 32));
            mx = fmaxf(mx, __shfl_xor(mx, 4, 32));
            mx = fmaxf(mx, __shfl_xor(mx, 8, 32));
            float mnew  = fmaxf(mi[r], mx);
            float scale = __expf(mi[r] - mnew);
            float p0    = __expf(x0 - mnew);
            float p1    = __expf(x1 - mnew);
            float rs = p0 + p1;
            rs += __shfl_xor(rs, 1, 32);
            rs += __shfl_xor(rs, 2, 32);
            rs += __shfl_xor(rs, 4, 32);
            rs += __shfl_xor(rs, 8, 32);
            li[r] = li[r] * scale + rs;
            mi[r] = mnew;
            o0[r] *= scale; o1[r] *= scale; o2[r] *= scale; o3[r] *= scale;
            s0[r] = p0; s1[r] = p1;
        }

        // transpose P (C-layout) -> A-fragment (16x32 bf16) via LDS
        __syncthreads();
#pragma unroll
        for (int r = 0; r < 8; ++r) {
            lds[r + rowoff][cl]      = s0[r];
            lds[r + rowoff][16 + cl] = s1[r];
        }
        __syncthreads();
        FragU pf;
        int kb8 = (lane >> 4) * 8;
#pragma unroll
        for (int t = 0; t < 8; ++t) {
            pf.s[t]     = f2bf(lds[cl][kb8 + t]);
            pf.s[8 + t] = f2bf(lds[cl][16 + kb8 + t]);
        }
        v16bf pa = pf.v;

        // O (16x64) += P (16x32) @ V_block (32x64); V stored [D][S]
        v16bf vb0 = load_frag_b(vv, 0,  j, SS);
        v16bf vb1 = load_frag_b(vv, 16, j, SS);
        v16bf vb2 = load_frag_b(vv, 32, j, SS);
        v16bf vb3 = load_frag_b(vv, 48, j, SS);
        o0 = wmma_bf16(pa, vb0, o0);
        o1 = wmma_bf16(pa, vb1, o1);
        o2 = wmma_bf16(pa, vb2, o2);
        o3 = wmma_bf16(pa, vb3, o3);
    }

    // epilogue: normalize, store bf16 into attn-out [B*S][H*D]
    int b = bh / HH, h = bh - (bh / HH) * HH;
#pragma unroll
    for (int r = 0; r < 8; ++r) {
        float inv  = 1.0f / li[r];
        int   srow = q0 + r + rowoff;
        size_t mrow = (size_t)(b * SS + srow) * DM + (size_t)h * DH;
        ao[mrow +  0 + cl] = f2bf(o0[r] * inv);
        ao[mrow + 16 + cl] = f2bf(o1[r] * inv);
        ao[mrow + 32 + cl] = f2bf(o2[r] * inv);
        ao[mrow + 48 + cl] = f2bf(o3[r] * inv);
    }
}

// ---------------- output projection GEMM ----------------
// out[m][n] = sum_k ao[m][k] * WoT[n][k];  M=4096, N=1024, K=1024 (fp32 out)
// One wave per 32x64 tile.
__global__ __launch_bounds__(32)
void k_gemm_out(const unsigned short* __restrict__ ao,
                const unsigned short* __restrict__ woT,
                float* __restrict__ out) {
    int nt = blockIdx.x;
    int mt = blockIdx.y;
    int n0 = nt * 64;
    int m0 = mt * 32;
    v8f acc[2][4] = {};
    for (int k = 0; k < DM; k += 32) {
        v16bf a0 = load_frag_a(ao, m0,      DM, k);
        v16bf a1 = load_frag_a(ao, m0 + 16, DM, k);
        v16bf b0 = load_frag_b(woT, n0,      k, DM);
        v16bf b1 = load_frag_b(woT, n0 + 16, k, DM);
        v16bf b2 = load_frag_b(woT, n0 + 32, k, DM);
        v16bf b3 = load_frag_b(woT, n0 + 48, k, DM);
        acc[0][0] = wmma_bf16(a0, b0, acc[0][0]);
        acc[0][1] = wmma_bf16(a0, b1, acc[0][1]);
        acc[0][2] = wmma_bf16(a0, b2, acc[0][2]);
        acc[0][3] = wmma_bf16(a0, b3, acc[0][3]);
        acc[1][0] = wmma_bf16(a1, b0, acc[1][0]);
        acc[1][1] = wmma_bf16(a1, b1, acc[1][1]);
        acc[1][2] = wmma_bf16(a1, b2, acc[1][2]);
        acc[1][3] = wmma_bf16(a1, b3, acc[1][3]);
    }
    int lane   = threadIdx.x & 31;
    int cl     = lane & 15;
    int rowoff = (lane >> 4) * 8;
#pragma unroll
    for (int t = 0; t < 2; ++t)
#pragma unroll
        for (int c = 0; c < 4; ++c)
#pragma unroll
            for (int r = 0; r < 8; ++r)
                out[(size_t)(m0 + t * 16 + r + rowoff) * DM + n0 + c * 16 + cl] =
                    acc[t][c][r];
}

// ---------------- launcher ----------------
extern "C" void kernel_launch(void* const* d_in, const int* in_sizes, int n_in,
                              void* d_out, int out_size, void* d_ws, size_t ws_size,
                              hipStream_t stream) {
    (void)in_sizes; (void)n_in; (void)out_size; (void)ws_size;
    const float* x    = (const float*)d_in[0];
    const float* wqkv = (const float*)d_in[1];
    const float* wo   = (const float*)d_in[2];
    // d_in[3] = causal mask: recomputed analytically in-kernel
    float* out = (float*)d_out;

    char* ws = (char*)d_ws;
    size_t off = 0;
    unsigned short* xb    = (unsigned short*)(ws + off); off += (size_t)MTOT * DM * 2;           // 8 MB
    unsigned short* wqkvT = (unsigned short*)(ws + off); off += (size_t)NQKV * DM * 2;           // 6 MB
    unsigned short* woT   = (unsigned short*)(ws + off); off += (size_t)DM * DM * 2;             // 2 MB
    unsigned short* qb    = (unsigned short*)(ws + off); off += (size_t)BB * HH * SS * DH * 2;   // 8 MB
    unsigned short* kb    = (unsigned short*)(ws + off); off += (size_t)BB * HH * SS * DH * 2;   // 8 MB
    unsigned short* vb    = (unsigned short*)(ws + off); off += (size_t)BB * HH * SS * DH * 2;   // 8 MB
    unsigned short* ao    = (unsigned short*)(ws + off); off += (size_t)MTOT * DM * 2;           // 8 MB

    k_pack_x<<<(MTOT * DM) / 256, 256, 0, stream>>>(x, xb, MTOT * DM);
    k_pack_wT<<<(NQKV * DM) / 256, 256, 0, stream>>>(wqkv, wqkvT, DM, NQKV);
    k_pack_wT<<<(DM * DM) / 256, 256, 0, stream>>>(wo, woT, DM, DM);

    k_gemm_qkv<<<dim3(NQKV / 64, MTOT / 32), 32, 0, stream>>>(xb, wqkvT, qb, kb, vb);
    k_attn<<<dim3(SS / 16, BB * HH), 32, 0, stream>>>(qb, kb, vb, ao);
    k_gemm_out<<<dim3(DM / 64, MTOT / 32), 32, 0, stream>>>(ao, woT, out);
}
